// Decoder_16698832846970
// MI455X (gfx1250) — matrix operand order repaired and checked by reading, not compiled, not run on hardware
//
#include <hip/hip_runtime.h>
#include <hip/hip_bf16.h>
#include <math.h>

typedef __attribute__((ext_vector_type(16))) __bf16 v16bf;
typedef __attribute__((ext_vector_type(8)))  float  v8f;

#define GHW 224
#define NPIX (GHW*GHW)          // 50176
#define NB 8
#define NCH 32

// ---------------------------------------------------------------- utilities
__device__ __forceinline__ int reflect224(int i) {
    i = i < 0 ? -i : i;
    return i >= GHW ? (2*GHW - 2 - i) : i;
}

__device__ __forceinline__ float cubicw(float d) {
    const float a = -0.75f;
    d = fabsf(d);
    float d2 = d*d, d3 = d2*d;
    if (d <= 1.f) return (a + 2.f)*d3 - (a + 3.f)*d2 + 1.f;
    if (d <  2.f) return a*d3 - 5.f*a*d2 + 8.f*a*d - 4.f*a;
    return 0.f;
}

// ------------------------------------------------------- 1) linear 1000->972
__global__ __launch_bounds__(256) void linear_kernel(
    const float* __restrict__ x, const float* __restrict__ W,
    const float* __restrict__ bias, float* __restrict__ f0)
{
    int i = blockIdx.x * 256 + threadIdx.x;
    if (i >= NB * 972) return;
    int b = i / 972, j = i % 972;
    const float* xr = x + b * 1000;
    const float* wr = W + j * 1000;
    float s = bias[j];
    for (int k = 0; k < 1000; ++k) s = fmaf(xr[k], wr[k], s);
    f0[i] = s;
}

// ------------------------------------------------- 2) bicubic 18x18 -> 224^2
__device__ __forceinline__ void bicubic_taps(int o, float* w, int* idx) {
    float xx = (o + 0.5f) * (18.0f / 224.0f) - 0.5f;
    float xf = floorf(xx);
    int x0 = (int)xf;
    float t = xx - xf;
    #pragma unroll
    for (int k = -1; k <= 2; ++k) {
        w[k + 1] = cubicw(t - (float)k);
        int id = x0 + k;
        idx[k + 1] = id < 0 ? 0 : (id > 17 ? 17 : id);
    }
}

__global__ __launch_bounds__(256) void bicubic_kernel(
    const float* __restrict__ f0, float* __restrict__ out)
{
    int i = blockIdx.x * 256 + threadIdx.x;
    if (i >= NB * 3 * NPIX) return;
    int ox = i % GHW; int r = i / GHW; int oy = r % GHW; int bc = r / GHW;
    float wy[4], wx[4]; int iy[4], ix[4];
    bicubic_taps(oy, wy, iy);
    bicubic_taps(ox, wx, ix);
    const float* src = f0 + bc * 324;   // 18*18
    float s = 0.f;
    #pragma unroll
    for (int ky = 0; ky < 4; ++ky) {
        const float* row = src + iy[ky] * 18;
        float rs = 0.f;
        #pragma unroll
        for (int kx = 0; kx < 4; ++kx) rs = fmaf(wx[kx], row[ix[kx]], rs);
        s = fmaf(wy[ky], rs, s);
    }
    out[i] = s;
}

// ------------------------- 3) range projection: q = W2*gelu(W1*g+b1)+b2 (WMMA)
// One wave computes 32 channels x 16 pixels via bf16x3-split WMMA (f32 accum).
__global__ __launch_bounds__(256) void proj_kernel(
    const float* __restrict__ g,  const float* __restrict__ w1,
    const float* __restrict__ b1, const float* __restrict__ w2,
    const float* __restrict__ b2, float* __restrict__ qout)
{
    __shared__ float w2l[1024];
    __shared__ float w1l[96];
    __shared__ float b1l[32];
    __shared__ float b2l[32];
    __shared__ float gt[8][3][16];
    __shared__ float Hl[8][32 * 18];      // H[c][pix], stride 18 (bank-spread)

    int tid = threadIdx.x;
    for (int i = tid; i < 1024; i += 256) w2l[i] = w2[i];
    if (tid < 96) w1l[tid] = w1[tid];
    if (tid < 32) { b1l[tid] = b1[tid]; b2l[tid] = b2[tid]; }
    __syncthreads();

    int wv = tid >> 5, lane = tid & 31;
    int tile = blockIdx.x * 8 + wv;            // 16 pixels per tile
    int pixBase = tile * 16;
    int b   = pixBase / NPIX;                  // tiles never cross batch (50176%16==0)
    int pib = pixBase % NPIX;

    // stage guidance for 16 pixels x 3 channels into LDS
    for (int idx = lane; idx < 48; idx += 32) {
        int c = idx >> 4, j = idx & 15;
        gt[wv][c][j] = g[(b * 3 + c) * NPIX + pib + j];
    }
    __syncthreads();

    // lane computes hidden channel c=lane for all 16 pixels
    {
        float a0 = w1l[lane * 3 + 0], a1 = w1l[lane * 3 + 1];
        float a2 = w1l[lane * 3 + 2], bb = b1l[lane];
        #pragma unroll
        for (int j = 0; j < 16; ++j) {
            float h = fmaf(a0, gt[wv][0][j], fmaf(a1, gt[wv][1][j], fmaf(a2, gt[wv][2][j], bb)));
            h = 0.5f * h * (1.0f + erff(h * 0.70710678118654752f));   // exact GELU
            Hl[wv][lane * 18 + j] = h;
        }
    }
    __syncthreads();

    // gather WMMA fragments per CDNA5 16-bit layouts
    int half = lane >> 4, mn = lane & 15;
    v16bf aH0, aL0, aH1, aL1, bH, bL;
    #pragma unroll
    for (int j = 0; j < 16; ++j) {
        int v = j >> 1, e = j & 1;
        int kk = (v < 4 ? 2 * v : 16 + 2 * (v - 4)) + e + 8 * half;  // A: K index
        float wA = w2l[mn * 32 + kk];          // A0 rows 0..15  (out ch)
        float wB = w2l[(16 + mn) * 32 + kk];   // A1 rows 16..31
        __bf16 h;
        h = (__bf16)wA; aH0[j] = h; aL0[j] = (__bf16)(wA - (float)h);
        h = (__bf16)wB; aH1[j] = h; aL1[j] = (__bf16)(wB - (float)h);
        int kb = (j < 8 ? j : j + 8) + 8 * half;  // B: K index
        float hb = Hl[wv][kb * 18 + mn];       // B[K][n=pixel]
        h = (__bf16)hb; bH[j]  = h; bL[j]  = (__bf16)(hb - (float)h);
    }

    v8f acc0, acc1;
    #pragma unroll
    for (int r = 0; r < 8; ++r) {
        acc0[r] = b2l[r + 8 * half];           // D element (r,lane): M = r+8*half
        acc1[r] = b2l[16 + r + 8 * half];
    }
    // bf16x3 split: D = b2 + Alo*Bhi + Ahi*Blo + Ahi*Bhi  (~f32 accuracy)
    acc0 = __builtin_amdgcn_wmma_f32_16x16x32_bf16(false, aL0, false, bH, (short)0, acc0, false, false);
    acc0 = __builtin_amdgcn_wmma_f32_16x16x32_bf16(false, aH0, false, bL, (short)0, acc0, false, false);
    acc0 = __builtin_amdgcn_wmma_f32_16x16x32_bf16(false, aH0, false, bH, (short)0, acc0, false, false);
    acc1 = __builtin_amdgcn_wmma_f32_16x16x32_bf16(false, aL1, false, bH, (short)0, acc1, false, false);
    acc1 = __builtin_amdgcn_wmma_f32_16x16x32_bf16(false, aH1, false, bL, (short)0, acc1, false, false);
    acc1 = __builtin_amdgcn_wmma_f32_16x16x32_bf16(false, aH1, false, bH, (short)0, acc1, false, false);

    #pragma unroll
    for (int r = 0; r < 8; ++r) {
        int c0 = r + 8 * half;
        qout[(b * NCH + c0     ) * NPIX + pib + mn] = acc0[r];
        qout[(b * NCH + c0 + 16) * NPIX + pib + mn] = acc1[r];
    }
}

// ---- 4) fused JBU: scores -> softmax*spatial -> normalize -> adaptive conv
#define TW 16
#define THH 8
#define PW 22          // TW + 6
#define PH 14          // THH + 6
#define QS 36          // padded channel stride (16B aligned, bank-spread)

__global__ __launch_bounds__(128) void jbu_kernel(
    const float* __restrict__ src, const float* __restrict__ q,
    const float* __restrict__ tempp, const float* __restrict__ sigp,
    float* __restrict__ out)
{
    __shared__ float qs[PH * PW * QS];
    __shared__ float hrs[3][PH][PW];

    int tid = threadIdx.x;
    int b   = blockIdx.z;
    int bx0 = blockIdx.x * TW;
    int by0 = blockIdx.y * THH;

    // stage q patch [PH x PW x 32] (reflect borders)
    for (int i = tid; i < NCH * PH * PW; i += 128) {
        int xp = i % PW; int r = i / PW; int yp = r % PH; int c = r / PH;
        int gy = reflect224(by0 + yp - 3);
        int gx = reflect224(bx0 + xp - 3);
        qs[(yp * PW + xp) * QS + c] = q[((b * NCH + c) * GHW + gy) * GHW + gx];
    }
    // stage hr patch [3 x PH x PW]
    for (int i = tid; i < 3 * PH * PW; i += 128) {
        int xp = i % PW; int r = i / PW; int yp = r % PH; int c = r / PH;
        int gy = reflect224(by0 + yp - 3);
        int gx = reflect224(bx0 + xp - 3);
        hrs[c][yp][xp] = src[((b * 3 + c) * GHW + gy) * GHW + gx];
    }
    __syncthreads();

    int tx = tid & 15, ty = tid >> 4;
    const float* qc = &qs[((ty + 3) * PW + (tx + 3)) * QS];
    float4 qc4[8];
    #pragma unroll
    for (int i = 0; i < 8; ++i) qc4[i] = *(const float4*)(qc + 4 * i);

    float sc[49];
    #pragma unroll
    for (int di = 0; di < 7; ++di) {
        #pragma unroll
        for (int dj = 0; dj < 7; ++dj) {
            const float* qo = &qs[((ty + di) * PW + (tx + dj)) * QS];
            float s0 = 0.f, s1 = 0.f, s2 = 0.f, s3 = 0.f;
            #pragma unroll
            for (int i = 0; i < 8; ++i) {
                float4 v = *(const float4*)(qo + 4 * i);
                s0 = fmaf(v.x, qc4[i].x, s0);
                s1 = fmaf(v.y, qc4[i].y, s1);
                s2 = fmaf(v.z, qc4[i].z, s2);
                s3 = fmaf(v.w, qc4[i].w, s3);
            }
            sc[di * 7 + dj] = (s0 + s1) + (s2 + s3);
        }
    }

    float t = expf(*tempp);
    t = fminf(fmaxf(t, 1e-4f), 1e4f);
    float sig = *sigp;
    float inv2s2 = 1.0f / (2.0f * sig * sig);

    float m = -1e30f;
    #pragma unroll
    for (int p = 0; p < 49; ++p) m = fmaxf(m, t * sc[p]);

    float sum = 0.f, o0 = 0.f, o1 = 0.f, o2 = 0.f;
    #pragma unroll
    for (int di = 0; di < 7; ++di) {
        float dy = (float)(di - 3) * (1.0f / 3.0f);
        #pragma unroll
        for (int dj = 0; dj < 7; ++dj) {
            float dx = (float)(dj - 3) * (1.0f / 3.0f);
            float w = expf(t * sc[di * 7 + dj] - m) * expf(-(dy * dy + dx * dx) * inv2s2);
            sum += w;
            o0 = fmaf(w, hrs[0][ty + di][tx + dj], o0);
            o1 = fmaf(w, hrs[1][ty + di][tx + dj], o1);
            o2 = fmaf(w, hrs[2][ty + di][tx + dj], o2);
        }
    }
    float inv = 1.0f / fmaxf(sum, 1e-30f);
    int oy = by0 + ty, ox = bx0 + tx;
    out[((b * 3 + 0) * GHW + oy) * GHW + ox] = o0 * inv;
    out[((b * 3 + 1) * GHW + oy) * GHW + ox] = o1 * inv;
    out[((b * 3 + 2) * GHW + oy) * GHW + ox] = o2 * inv;
}

// --------------------------------------------------------------------- launch
extern "C" void kernel_launch(void* const* d_in, const int* in_sizes, int n_in,
                              void* d_out, int out_size, void* d_ws, size_t ws_size,
                              hipStream_t stream)
{
    const float* x        = (const float*)d_in[0];
    const float* guidance = (const float*)d_in[1];
    const float* linear_w = (const float*)d_in[2];
    const float* linear_b = (const float*)d_in[3];
    const float* w1s      = (const float*)d_in[4];
    const float* b1s      = (const float*)d_in[5];
    const float* w2s      = (const float*)d_in[6];
    const float* b2s      = (const float*)d_in[7];
    const float* temps    = (const float*)d_in[8];
    const float* sigmas   = (const float*)d_in[9];

    char* ws = (char*)d_ws;
    const size_t srcBytes = (size_t)NB * 3 * NPIX * sizeof(float);  // 4,816,896
    float* f0   = (float*)ws;                                  // 7,776 floats
    float* srcA = (float*)(ws + 32768);
    float* srcB = (float*)(ws + 32768 + srcBytes);
    float* qbuf = (float*)(ws + 32768 + 2 * srcBytes);         // 8*32*50176 floats

    linear_kernel <<<(NB * 972 + 255) / 256, 256, 0, stream>>>(x, linear_w, linear_b, f0);
    bicubic_kernel<<<(NB * 3 * NPIX + 255) / 256, 256, 0, stream>>>(f0, srcA);

    float* srcp = srcA;
    float* dstp = srcB;
    for (int s = 0; s < 4; ++s) {
        proj_kernel<<<(NB * NPIX) / (8 * 16), 256, 0, stream>>>(
            guidance, w1s + s * 96, b1s + s * 32, w2s + s * 1024, b2s + s * 32, qbuf);
        float* outp = (s == 3) ? (float*)d_out : dstp;
        jbu_kernel<<<dim3(GHW / TW, GHW / THH, NB), 128, 0, stream>>>(
            srcp, qbuf, temps + s, sigmas + s, outp);
        float* tswap = srcp; srcp = dstp; dstp = tswap;
    }
}
